// SparseGraphNetwork_50654844289544
// MI455X (gfx1250) — compile-verified, matrix operand order checked by reading, not run on hardware
//
#include <hip/hip_runtime.h>
#include <math.h>

#define NNODE 2000
#define NPAD  2048       // node rows padded to a multiple of 64 for the GEMM
#define EDG   40000
#define GAM   20
#define DCH   128
#define NL    30
#define RB    4          // 16-row subtiles per wave in the GEMM (compile-time)
#define CCONST 10.0f

typedef __attribute__((ext_vector_type(16))) __bf16 v16bf;
typedef __attribute__((ext_vector_type(8)))  float  v8f;

// ---- helpers ---------------------------------------------------------------
__device__ __forceinline__ __bf16 f2bf(float x) {
  unsigned u = __float_as_uint(x);
  unsigned r = (u + 0x7FFFu + ((u >> 16) & 1u)) >> 16;   // round-to-nearest-even
  return __builtin_bit_cast(__bf16, (unsigned short)r);
}

// ---- weight prep: W[mats][k][n] (f32) -> Wt[mats][n][k] (bf16, transposed) --
__global__ void cvt_w_kernel(const float* __restrict__ W, __bf16* __restrict__ Wt, int mats) {
  int idx = blockIdx.x * blockDim.x + threadIdx.x;
  int total = mats * DCH * DCH;
  if (idx >= total) return;
  int m = idx >> 14;
  int n = (idx >> 7) & 127;
  int k = idx & 127;
  Wt[idx] = f2bf(W[(m << 14) + (k << 7) + n]);
}

// ---- WMMA GEMM: Y[M,128] = X16[M,128] @ W  (all operands bf16 in memory) ----
// Requires M % 64 == 0 (callers pad). Each wave: 64x16 output strip = RB
// row-subtiles sharing one register-resident B fragment per k-step.
// Straight-line: no divergent guards -> EXEC all-ones for every WMMA.
// flags: bit0 = add bias, bit1 = relu, bit2 = also write bf16 copy to Y16
__global__ void gemm128_bf16(const __bf16* __restrict__ X16, const __bf16* __restrict__ Wt,
                             const float* __restrict__ bias, float* __restrict__ Y,
                             __bf16* __restrict__ Y16, int M, int flags) {
  int wavesPerBlock = blockDim.x >> 5;
  int wave = blockIdx.x * wavesPerBlock + (threadIdx.x >> 5);
  int lane = threadIdx.x & 31;
  int tiles = (M >> 6) * 8;               // 64-row strips x (128/16) column tiles
  if (wave >= tiles) return;              // wave-uniform in practice (exact grids)
  int tmb = wave >> 3, tn = wave & 7;
  int r  = lane & 15, hi = lane >> 4;
  const __bf16* wrow  = Wt  + (size_t)(tn * 16 + r) * DCH;
  const __bf16* xbase = X16 + (size_t)(tmb * 64 + r) * DCH;
  __builtin_prefetch(xbase, 0, 0);        // global_prefetch_b8
  v8f acc[RB] = {};
#pragma unroll
  for (int k0 = 0; k0 < DCH; k0 += 32) {
    v16bf b;
#pragma unroll
    for (int j = 0; j < 8; ++j) {
      // 16-bit B 32x16 layout: lane half hi selects K groups {8*hi..} and {16+8*hi..}
      b[j]     = wrow[k0 + 8 * hi + j];
      b[j + 8] = wrow[k0 + 16 + 8 * hi + j];
    }
#pragma unroll
    for (int s = 0; s < RB; ++s) {
      const __bf16* xrow = xbase + (size_t)s * 16 * DCH;
      v16bf a;
#pragma unroll
      for (int j = 0; j < 8; ++j) {
        a[j]     = xrow[k0 + 8 * hi + j];
        a[j + 8] = xrow[k0 + 16 + 8 * hi + j];
      }
      acc[s] = __builtin_amdgcn_wmma_f32_16x16x32_bf16(false, a, false, b,
                                                       (short)0, acc[s], false, false);
    }
  }
  int col = tn * 16 + r;
  float bv = (flags & 1) ? bias[col] : 0.0f;
#pragma unroll
  for (int s = 0; s < RB; ++s) {
#pragma unroll
    for (int j = 0; j < 8; ++j) {
      int row = tmb * 64 + s * 16 + 8 * hi + j;  // C/D layout: VGPR j -> M=j (+8 hi)
      float yv = acc[s][j] + bv;
      if (flags & 2) yv = fmaxf(yv, 0.0f);
      Y[(size_t)row * DCH + col] = yv;
      if (flags & 4) Y16[(size_t)row * DCH + col] = f2bf(yv);
    }
  }
}

// ---- encoders --------------------------------------------------------------
__global__ void edge_dist_kernel(const float* __restrict__ coords, const int* __restrict__ tgt,
                                 float* __restrict__ dist) {
  int k = blockIdx.x * blockDim.x + threadIdx.x;
  if (k >= EDG) return;
  int s = k / GAM, t = tgt[k];
  float dx = coords[2 * s] - coords[2 * t];
  float dy = coords[2 * s + 1] - coords[2 * t + 1];
  dist[k] = sqrtf(dx * dx + dy * dy);
}

__global__ void edge_init_kernel(const float* __restrict__ dist, const float* __restrict__ eW,
                                 const float* __restrict__ eb, float* __restrict__ e,
                                 __bf16* __restrict__ e16) {
  int idx = blockIdx.x * blockDim.x + threadIdx.x;
  if (idx >= EDG * DCH) return;
  int k = idx >> 7, c = idx & 127;
  float t = dist[k] * eW[c] + eb[c];
  if (!isfinite(t)) t = 0.0f;
  e[idx] = t;
  e16[idx] = f2bf(t);
}

// writes NPAD rows: rows >= NNODE are zero (pad for the WMMA GEMM; never re-written)
__global__ void node_init_kernel(const float* __restrict__ coords, const float* __restrict__ vW,
                                 const float* __restrict__ vb, float* __restrict__ v,
                                 __bf16* __restrict__ v16) {
  int idx = blockIdx.x * blockDim.x + threadIdx.x;
  if (idx >= NPAD * DCH) return;
  int i = idx >> 7, c = idx & 127;
  float t = 0.0f;
  if (i < NNODE) {
    t = coords[2 * i] * vW[c] + coords[2 * i + 1] * vW[DCH + c] + vb[c];
    if (!isfinite(t)) t = 0.0f;
  }
  v[idx] = t;
  v16[idx] = f2bf(t);
}

// ---- attention + aggregation (segments are contiguous blocks of 20) --------
__global__ void attn_agg_kernel(const float* __restrict__ al, const float* __restrict__ vWn,
                                const float* __restrict__ vWs, const float* __restrict__ bs,
                                const int* __restrict__ tgt, float* __restrict__ v_up) {
  int i = blockIdx.x;        // node
  int d = threadIdx.x;       // channel
  __shared__ int stgt[GAM];
  if (d < GAM) stgt[d] = tgt[i * GAM + d];
  __syncthreads();
  const float* base = al + (size_t)i * GAM * DCH + d;
  float av[GAM];
  float m = -INFINITY;
#pragma unroll
  for (int j = 0; j < GAM; ++j) { av[j] = base[j * DCH]; m = fmaxf(m, av[j]); }
  float ex[GAM], s = 0.0f;
#pragma unroll
  for (int j = 0; j < GAM; ++j) { ex[j] = __expf(av[j] - m); s += ex[j]; }
  s = fmaxf(s, 1e-10f);
  float inv = 1.0f / (s + 1e-10f);
  float agg = 0.0f;
#pragma unroll
  for (int j = 0; j < GAM; ++j) {
    float a = ex[j] * inv;
    if (!isfinite(a)) a = 0.0f;
    agg += a * vWn[(size_t)stgt[j] * DCH + d];
  }
  v_up[(size_t)i * DCH + d] = vWs[(size_t)i * DCH + d] + bs[d] + agg;
}

// ---- per-channel batch stats (mean, biased var) -----------------------------
__global__ void colstats_kernel(const float* __restrict__ X, int M, float* __restrict__ stats) {
  int c = blockIdx.x, t = threadIdx.x;
  float s1 = 0.0f, s2 = 0.0f;
  for (int r = t; r < M; r += blockDim.x) {
    float x = X[(size_t)r * DCH + c];
    s1 += x; s2 += x * x;
  }
  __shared__ float a1[256], a2[256];
  a1[t] = s1; a2[t] = s2; __syncthreads();
  for (int o = 128; o > 0; o >>= 1) {
    if (t < o) { a1[t] += a1[t + o]; a2[t] += a2[t + o]; }
    __syncthreads();
  }
  if (t == 0) {
    float mu = a1[0] / (float)M;
    stats[c] = mu;
    stats[DCH + c] = a2[0] / (float)M - mu * mu;
  }
}

// ---- BN + relu + residual (in place on x, refresh bf16 mirror) --------------
__global__ void bn_res_kernel(const float* __restrict__ xup, float* __restrict__ x,
                              __bf16* __restrict__ x16,
                              const float* __restrict__ stats, const float* __restrict__ g,
                              const float* __restrict__ b, int M) {
  int idx = blockIdx.x * blockDim.x + threadIdx.x;
  if (idx >= M * DCH) return;
  int c = idx & 127;
  float mu = stats[c], var = stats[DCH + c];
  float xn = g[c] * (xup[idx] - mu) * rsqrtf(var + 1e-5f) + b[c];
  float t = x[idx] + fmaxf(xn, 0.0f);
  if (!isfinite(t)) t = 0.0f;
  x[idx] = t;
  x16[idx] = f2bf(t);
}

// ---- r_miss = p @ Wr + br ---------------------------------------------------
__global__ void rmiss_kernel(const float* __restrict__ p, const float* __restrict__ Wr,
                             const float* __restrict__ br, float* __restrict__ out) {
  int d = threadIdx.x;
  float s = 0.0f;
  for (int k = 0; k < DCH; ++k) s += p[k] * Wr[k * DCH + d];
  out[d] = s + br[d];
}

// ---- edge update ------------------------------------------------------------
__global__ void edge_update_kernel(const float* __restrict__ vWf, const float* __restrict__ vWt,
                                   const float* __restrict__ eWo, const float* __restrict__ eWr,
                                   const int* __restrict__ tgt, const int* __restrict__ rev,
                                   const int* __restrict__ rex, const float* __restrict__ bf,
                                   const float* __restrict__ bt, const float* __restrict__ bo,
                                   const float* __restrict__ br, const float* __restrict__ rmiss,
                                   float* __restrict__ eup) {
  int idx = blockIdx.x * blockDim.x + threadIdx.x;
  if (idx >= EDG * DCH) return;
  int k = idx >> 7, c = idx & 127;
  int s = k / GAM;
  float r = rex[k] ? (eWr[(size_t)rev[k] * DCH + c] + br[c]) : rmiss[c];
  eup[idx] = vWf[(size_t)s * DCH + c] + bf[c]
           + vWt[(size_t)tgt[k] * DCH + c] + bt[c]
           + eWo[idx] + bo[c] + r;
}

// ---- decoder tails ----------------------------------------------------------
__global__ void rowdot_kernel(const float* __restrict__ X, const float* __restrict__ w,
                              float* __restrict__ out, int M) {
  int r = blockIdx.x * blockDim.x + threadIdx.x;
  if (r >= M) return;
  const float* x = X + (size_t)r * DCH;
  float s = 0.0f;
#pragma unroll
  for (int j = 0; j < DCH; ++j) s += x[j] * w[j];
  out[r] = s;
}

__global__ void beta_kernel(const float* __restrict__ bl, float* __restrict__ beta) {
  int i = blockIdx.x * blockDim.x + threadIdx.x;
  if (i >= NNODE) return;
  const float* b = bl + i * GAM;
  float m = -INFINITY;
#pragma unroll
  for (int j = 0; j < GAM; ++j) m = fmaxf(m, b[j]);
  float ex[GAM], s = 0.0f;
#pragma unroll
  for (int j = 0; j < GAM; ++j) {
    float z = fminf(fmaxf(b[j] - m, -20.0f), 20.0f);
    ex[j] = __expf(z); s += ex[j];
  }
  float inv = 1.0f / (s + 1e-10f);
#pragma unroll
  for (int j = 0; j < GAM; ++j) {
    float v = ex[j] * inv;
    if (!isfinite(v)) v = 1.0f / (float)EDG;
    beta[i * GAM + j] = v;
  }
}

__global__ void pi_kernel(const float* __restrict__ dot, float* __restrict__ out) {
  int r = blockIdx.x * blockDim.x + threadIdx.x;
  if (r >= NNODE) return;
  float t = CCONST * tanhf(dot[r]);
  if (!isfinite(t)) t = 0.0f;
  out[r] = t;
}

__global__ void finalize_kernel(const int* __restrict__ ei, const float* __restrict__ dist,
                                float* __restrict__ out_ei, float* __restrict__ out_dist) {
  int idx = blockIdx.x * blockDim.x + threadIdx.x;
  if (idx < 2 * EDG) out_ei[idx] = (float)ei[idx];
  if (idx < EDG)     out_dist[idx] = dist[idx];
}

// ---- host ------------------------------------------------------------------
// input order: setup_inputs() dict order, params flattened in insertion order
enum {
  I_COORDS = 0, I_EI, I_REX, I_RIX,
  I_ENC_VW, I_ENC_VB, I_ENC_EW, I_ENC_EB,
  I_WA, I_WSM, I_BS, I_WN, I_WF, I_BF, I_WTM, I_BT,
  I_WO, I_BO, I_WR, I_BR, I_P, I_BN_VG, I_BN_VB, I_BN_EG, I_BN_EB,
  I_ED1W, I_ED1B, I_ED2W, I_ED2B, I_WBETA,
  I_ND1W, I_ND1B, I_ND2W, I_ND2B, I_WPI
};

extern "C" void kernel_launch(void* const* d_in, const int* in_sizes, int n_in,
                              void* d_out, int out_size, void* d_ws, size_t ws_size,
                              hipStream_t stream) {
  (void)in_sizes; (void)n_in; (void)out_size; (void)ws_size;
  const float* coords = (const float*)d_in[I_COORDS];
  const int*   ei     = (const int*)d_in[I_EI];
  const int*   tgt    = ei + EDG;                 // edge_index[1]
  const int*   rex    = (const int*)d_in[I_REX];
  const int*   rix    = (const int*)d_in[I_RIX];

  const float* enc_vW = (const float*)d_in[I_ENC_VW];
  const float* enc_vb = (const float*)d_in[I_ENC_VB];
  const float* enc_eW = (const float*)d_in[I_ENC_EW];
  const float* enc_eb = (const float*)d_in[I_ENC_EB];

  const float* Wa = (const float*)d_in[I_WA];
  const float* Ws = (const float*)d_in[I_WSM];
  const float* bs = (const float*)d_in[I_BS];
  const float* Wn = (const float*)d_in[I_WN];
  const float* Wf = (const float*)d_in[I_WF];
  const float* bf = (const float*)d_in[I_BF];
  const float* Wt = (const float*)d_in[I_WTM];
  const float* bt = (const float*)d_in[I_BT];
  const float* Wo = (const float*)d_in[I_WO];
  const float* bo = (const float*)d_in[I_BO];
  const float* Wr = (const float*)d_in[I_WR];
  const float* br = (const float*)d_in[I_BR];
  const float* pp = (const float*)d_in[I_P];
  const float* bn_vg = (const float*)d_in[I_BN_VG];
  const float* bn_vb = (const float*)d_in[I_BN_VB];
  const float* bn_eg = (const float*)d_in[I_BN_EG];
  const float* bn_eb = (const float*)d_in[I_BN_EB];

  const float* ed1W = (const float*)d_in[I_ED1W];
  const float* ed1b = (const float*)d_in[I_ED1B];
  const float* ed2W = (const float*)d_in[I_ED2W];
  const float* ed2b = (const float*)d_in[I_ED2B];
  const float* Wbeta = (const float*)d_in[I_WBETA];
  const float* nd1W = (const float*)d_in[I_ND1W];
  const float* nd1b = (const float*)d_in[I_ND1B];
  const float* nd2W = (const float*)d_in[I_ND2W];
  const float* nd2b = (const float*)d_in[I_ND2B];
  const float* Wpi  = (const float*)d_in[I_WPI];

  float* out = (float*)d_out;

  // workspace bump allocator (everything read is written first -> deterministic)
  char* wsp = (char*)d_ws;
  auto alloc = [&](size_t bytes) -> void* {
    void* p = (void*)wsp;
    wsp += (bytes + 255) & ~(size_t)255;
    return p;
  };
  const size_t EF = (size_t)EDG * DCH, NF = (size_t)NPAD * DCH;  // node bufs padded
  float* e    = (float*)alloc(EF * 4);
  float* al   = (float*)alloc(EF * 4);   // also e_up / decoder hidden
  float* eWo  = (float*)alloc(EF * 4);   // also decoder e_f
  float* eWr  = (float*)alloc(EF * 4);
  float* v    = (float*)alloc(NF * 4);
  float* vWn  = (float*)alloc(NF * 4);
  float* vWs  = (float*)alloc(NF * 4);
  float* vWf  = (float*)alloc(NF * 4);
  float* vWt  = (float*)alloc(NF * 4);
  float* vup  = (float*)alloc(NF * 4);
  __bf16* e16 = (__bf16*)alloc(EF * 2);  // bf16 mirror of e
  __bf16* h16 = (__bf16*)alloc(EF * 2);  // bf16 decoder hidden (edge + node reuse)
  __bf16* v16 = (__bf16*)alloc(NF * 2);  // bf16 mirror of v (rows >= NNODE stay 0)
  float* dist = (float*)alloc((size_t)EDG * 4);
  float* bl   = (float*)alloc((size_t)EDG * 4);
  float* ndot = (float*)alloc((size_t)NNODE * 4);
  float* stats = (float*)alloc(2 * DCH * 4);
  float* rmiss = (float*)alloc(DCH * 4);
  const size_t WB = (size_t)NL * DCH * DCH;   // per-tensor layer-weight elems
  __bf16* WaT = (__bf16*)alloc(WB * 2);
  __bf16* WsT = (__bf16*)alloc(WB * 2);
  __bf16* WnT = (__bf16*)alloc(WB * 2);
  __bf16* WfT = (__bf16*)alloc(WB * 2);
  __bf16* WtT = (__bf16*)alloc(WB * 2);
  __bf16* WoT = (__bf16*)alloc(WB * 2);
  __bf16* WrT = (__bf16*)alloc(WB * 2);
  __bf16* ed1T = (__bf16*)alloc((size_t)DCH * DCH * 2);
  __bf16* ed2T = (__bf16*)alloc((size_t)DCH * DCH * 2);
  __bf16* nd1T = (__bf16*)alloc((size_t)DCH * DCH * 2);
  __bf16* nd2T = (__bf16*)alloc((size_t)DCH * DCH * 2);

  auto cdiv = [](int a, int b) { return (a + b - 1) / b; };
  auto gemm = [&](const __bf16* X16, const __bf16* W16, const float* bias, float* Y,
                  __bf16* Y16, int M, int flags) {   // M must be a multiple of 64
    int tiles = (M >> 6) * 8;
    gemm128_bf16<<<cdiv(tiles, 8), 256, 0, stream>>>(X16, W16, bias, Y, Y16, M, flags);
  };

  // 1) convert all GEMM weights to transposed bf16
  {
    int tl = NL * DCH * DCH, g = cdiv(tl, 256);
    cvt_w_kernel<<<g, 256, 0, stream>>>(Wa, WaT, NL);
    cvt_w_kernel<<<g, 256, 0, stream>>>(Ws, WsT, NL);
    cvt_w_kernel<<<g, 256, 0, stream>>>(Wn, WnT, NL);
    cvt_w_kernel<<<g, 256, 0, stream>>>(Wf, WfT, NL);
    cvt_w_kernel<<<g, 256, 0, stream>>>(Wt, WtT, NL);
    cvt_w_kernel<<<g, 256, 0, stream>>>(Wo, WoT, NL);
    cvt_w_kernel<<<g, 256, 0, stream>>>(Wr, WrT, NL);
    int g1 = cdiv(DCH * DCH, 256);
    cvt_w_kernel<<<g1, 256, 0, stream>>>(ed1W, ed1T, 1);
    cvt_w_kernel<<<g1, 256, 0, stream>>>(ed2W, ed2T, 1);
    cvt_w_kernel<<<g1, 256, 0, stream>>>(nd1W, nd1T, 1);
    cvt_w_kernel<<<g1, 256, 0, stream>>>(nd2W, nd2T, 1);
  }

  // 2) encoders (write f32 master + bf16 mirror; node pads zero rows to NPAD)
  edge_dist_kernel<<<cdiv(EDG, 256), 256, 0, stream>>>(coords, tgt, dist);
  edge_init_kernel<<<cdiv(EDG * DCH, 256), 256, 0, stream>>>(dist, enc_eW, enc_eb, e, e16);
  node_init_kernel<<<cdiv(NPAD * DCH, 256), 256, 0, stream>>>(coords, enc_vW, enc_vb, v, v16);

  // 3) layers
  for (int l = 0; l < NL; ++l) {
    size_t wo = (size_t)l * DCH * DCH;
    const float* bsl = bs + l * DCH;  const float* bfl = bf + l * DCH;
    const float* btl = bt + l * DCH;  const float* bol = bo + l * DCH;
    const float* brl = br + l * DCH;
    // GEMMs from pre-update v16/e16 mirrors (pure bf16 WMMA, no cvt in loop)
    gemm(e16, WaT + wo, nullptr, al,  nullptr, EDG,  0);
    gemm(e16, WoT + wo, nullptr, eWo, nullptr, EDG,  0);
    gemm(e16, WrT + wo, nullptr, eWr, nullptr, EDG,  0);
    gemm(v16, WnT + wo, nullptr, vWn, nullptr, NPAD, 0);
    gemm(v16, WsT + wo, nullptr, vWs, nullptr, NPAD, 0);
    gemm(v16, WfT + wo, nullptr, vWf, nullptr, NPAD, 0);
    gemm(v16, WtT + wo, nullptr, vWt, nullptr, NPAD, 0);
    // node side: attention softmax over contiguous 20-edge segments + aggregate
    attn_agg_kernel<<<NNODE, 128, 0, stream>>>(al, vWn, vWs, bsl, tgt, vup);
    colstats_kernel<<<DCH, 256, 0, stream>>>(vup, NNODE, stats);
    bn_res_kernel<<<cdiv(NNODE * DCH, 256), 256, 0, stream>>>(
        vup, v, v16, stats, bn_vg + l * DCH, bn_vb + l * DCH, NNODE);
    // edge side
    rmiss_kernel<<<1, 128, 0, stream>>>(pp + l * DCH, Wr + wo, brl, rmiss);
    edge_update_kernel<<<cdiv(EDG * DCH, 256), 256, 0, stream>>>(
        vWf, vWt, eWo, eWr, tgt, rix, rex, bfl, btl, bol, brl, rmiss, al);
    colstats_kernel<<<DCH, 256, 0, stream>>>(al, EDG, stats);
    bn_res_kernel<<<cdiv(EDG * DCH, 256), 256, 0, stream>>>(
        al, e, e16, stats, bn_eg + l * DCH, bn_eb + l * DCH, EDG);
  }

  // 4) edge decoder -> beta  (flags: 1=bias, 2=relu, 4=emit bf16 copy)
  gemm(e16, ed1T, ed1b, al,  h16,     EDG, 1 | 2 | 4);
  gemm(h16, ed2T, ed2b, eWo, nullptr, EDG, 1 | 2);
  rowdot_kernel<<<cdiv(EDG, 256), 256, 0, stream>>>(eWo, Wbeta, bl, EDG);
  beta_kernel<<<cdiv(NNODE, 256), 256, 0, stream>>>(bl, out);

  // 5) node decoder -> pi (padded rows flow through deterministically, unused)
  gemm(v16, nd1T, nd1b, vWn, h16,     NPAD, 1 | 2 | 4);
  gemm(h16, nd2T, nd2b, vWs, nullptr, NPAD, 1 | 2);
  rowdot_kernel<<<cdiv(NNODE, 256), 256, 0, stream>>>(vWs, Wpi, ndot, NNODE);
  pi_kernel<<<cdiv(NNODE, 256), 256, 0, stream>>>(ndot, out + EDG);

  // 6) pass-through outputs: edge_index (as float) and edge_dist
  finalize_kernel<<<cdiv(2 * EDG, 256), 256, 0, stream>>>(
      ei, dist, out + EDG + NNODE, out + EDG + NNODE + 2 * EDG);
}
